// CrossAttentionBlock_59450937311464
// MI455X (gfx1250) — matrix operand order repaired and checked by reading, not compiled, not run on hardware
//
#include <hip/hip_runtime.h>
#include <hip/hip_bf16.h>

// ---------------------------------------------------------------------------
// CrossAttentionBlock fused for MI455X (gfx1250, wave32, WMMA f16 16x16x32).
//
//   K1 gn_stats : per (b,group) mean/rstd of x
//   K2 kv_ln    : LayerNorm(context) -> f16, key rows padded to 96 (zeros)
//   K3 kv_gemm  : K [key][hd] (softmax scale folded), V^T [hd][key], f16
//   K4 wcvt     : Wq,Wo -> transposed f16 (WqT/WoT, N-major) via LDS transpose
//   K5 attn     : per 128-query tile, fused in 243 KB LDS:
//                 GN-normalize -> Q=A@Wq -> scores=Q@K^T -> softmax ->
//                 O=P@V -> out=O@Wo + bo + residual(x)
//
// All WMMA B-operands are kept N-major so each lane's 16 K-values are one
// 32-byte contiguous strip (2x b128 loads); A-fragments are 2x b128 loads.
// Weight chunks are staged into LDS with async load-to-LDS when available.
// ---------------------------------------------------------------------------

typedef __attribute__((ext_vector_type(16))) _Float16 v16h;
typedef __attribute__((ext_vector_type(8)))  _Float16 v8h;
typedef __attribute__((ext_vector_type(8)))  float    v8f;
typedef __attribute__((ext_vector_type(4)))  int      v4i_;

#define AS_GLOBAL __attribute__((address_space(1)))
#define AS_SHARED __attribute__((address_space(3)))

#define WMMA_F16(a, b, c) \
  __builtin_amdgcn_wmma_f32_16x16x32_f16(false, (a), false, (b), (short)0, (c), false, false)

#if defined(__has_builtin)
#if __has_builtin(__builtin_amdgcn_global_load_async_to_lds_b128) && \
    __has_builtin(__builtin_amdgcn_s_wait_asynccnt)
#define USE_ASYNC_STAGE 1
#endif
#endif

constexpr int   B_    = 2;
constexpr int   C_    = 256;
constexpr int   S_    = 32768;      // 32*32*32 spatial positions
constexpr int   G_    = 8;
constexpr int   CPG_  = C_ / G_;    // 32
constexpr int   H_    = 8;
constexpr int   HD_   = 32;
constexpr int   CTX_  = 77;
constexpr int   CTXD_ = 768;
constexpr int   KPAD_ = 96;         // keys padded to multiple of 32
constexpr int   MT_   = 128;        // query rows per workgroup (16 per wave)
constexpr int   LDA_  = C_ + 8;     // 264 halves (stride*2 % 16B == 0)
constexpr int   LDP_  = KPAD_ + 8;  // 104 halves
constexpr int   LDT_  = C_ + 8;     // 264 halves, staged transposed weights
constexpr float EPS_  = 1e-5f;
constexpr float SCALE_ = 0.17677669529663687f;  // 1/sqrt(32), folded into K

constexpr size_t SMEM_ATTN =
    (size_t)(MT_ * LDA_            // sA  (f16) xnorm tile, later attn-out
           + MT_ * LDA_            // sQ  (f16)
           + MT_ * LDP_            // sP  (f16) probabilities
           + 64 * LDT_) * 2        // sBt (f16) staged weight chunk, N-major
  + (size_t)(MT_ * LDP_) * 4;      // sS  (f32) raw scores
// = 248832 bytes < 320 KB per workgroup

// --------------------------- fragment helpers ------------------------------
// A (16x32 f16, MxK): lane m=lane&15, K-half by lane<16.
// a[0..15] = row[kb..kb+15], kb = 0 or 16  (two 16B loads)
static __device__ __forceinline__ v16h load_a_frag(const _Float16* A, int lda, int lane) {
  int m  = lane & 15;
  int kb = (lane < 16) ? 0 : 16;
  const _Float16* row = A + (size_t)m * lda + kb;
  v8h lo = *(const v8h*)(row);
  v8h hi = *(const v8h*)(row + 8);
  v16h a;
#pragma unroll
  for (int j = 0; j < 8; ++j) { a[j] = lo[j]; a[8 + j] = hi[j]; }
  return a;
}

// B (32x16 f16, KxN) from an N-major (transposed) buffer Bt[n][k], ldt halves.
// Lane n needs K = kb..kb+15 of its column: one contiguous 32B strip.
static __device__ __forceinline__ v16h load_b_frag_t(const _Float16* Bt, int ldt, int lane) {
  int n  = lane & 15;
  int kb = (lane < 16) ? 0 : 16;
  const _Float16* p = Bt + (size_t)n * ldt + kb;
  v8h lo = *(const v8h*)(p);
  v8h hi = *(const v8h*)(p + 8);
  v16h b;
#pragma unroll
  for (int j = 0; j < 8; ++j) { b[j] = lo[j]; b[8 + j] = hi[j]; }
  return b;
}

// B from row-major f32 (only used by the tiny K/V projection kernel).
static __device__ __forceinline__ v16h load_b_frag_f32(const float* Bm, int ldb, int lane) {
  int n  = lane & 15;
  int kb = (lane < 16) ? 0 : 16;
  v16h b;
#pragma unroll
  for (int j = 0; j < 8; ++j) {
    b[2 * j]     = (_Float16)Bm[(size_t)(kb + 2 * j) * ldb + n];
    b[2 * j + 1] = (_Float16)Bm[(size_t)(kb + 2 * j + 1) * ldb + n];
  }
  return b;
}

// C/D (16x16 f32): N = lane&15, M = (lane<16?0:8)+vgpr.
static __device__ __forceinline__ void store_d_f16(_Float16* D, int ldd, int lane, v8f acc) {
  int n  = lane & 15;
  int mb = (lane < 16) ? 0 : 8;
#pragma unroll
  for (int j = 0; j < 8; ++j) D[(size_t)(mb + j) * ldd + n] = (_Float16)acc[j];
}

#if USE_ASYNC_STAGE
// Async copy of one 16-byte chunk global->LDS (ASYNCcnt-tracked).
static __device__ __forceinline__ void async_copy16(const _Float16* gsrc, _Float16* ldst) {
  __builtin_amdgcn_global_load_async_to_lds_b128(
      (AS_GLOBAL v4i_*)(void*)(const_cast<_Float16*>(gsrc)),
      (AS_SHARED v4i_*)(void*)ldst, 0, 0);
}
#endif

// ------------------------------- kernel 1 ----------------------------------
__global__ __launch_bounds__(256) void gn_stats_kernel(const float* __restrict__ x,
                                                       float* __restrict__ gmean,
                                                       float* __restrict__ grstd) {
  __shared__ float red[512];
  int b = blockIdx.x >> 3, g = blockIdx.x & 7;
  const float* p = x + ((size_t)b * C_ + (size_t)g * CPG_) * S_;
  const int N = CPG_ * S_;
  float s = 0.f, sq = 0.f;
  const float4* p4 = (const float4*)p;
  for (int i = threadIdx.x; i < N / 4; i += 256) {
    float4 v = p4[i];
    s  += v.x + v.y + v.z + v.w;
    sq += v.x * v.x + v.y * v.y + v.z * v.z + v.w * v.w;
  }
  red[threadIdx.x] = s;
  red[256 + threadIdx.x] = sq;
  __syncthreads();
  for (int o = 128; o > 0; o >>= 1) {
    if (threadIdx.x < o) {
      red[threadIdx.x] += red[threadIdx.x + o];
      red[256 + threadIdx.x] += red[256 + threadIdx.x + o];
    }
    __syncthreads();
  }
  if (threadIdx.x == 0) {
    float mean = red[0] / (float)N;
    float var  = red[256] / (float)N - mean * mean;
    gmean[b * G_ + g] = mean;
    grstd[b * G_ + g] = rsqrtf(var + EPS_);
  }
}

// ------------------------------- kernel 2 ----------------------------------
__global__ __launch_bounds__(256) void kv_ln_kernel(const float* __restrict__ ctx,
                                                    const float* __restrict__ lng,
                                                    const float* __restrict__ lnb,
                                                    _Float16* __restrict__ ctx16) {
  __shared__ float red[512];
  int b = blockIdx.x / KPAD_, r = blockIdx.x % KPAD_;
  _Float16* orow = ctx16 + ((size_t)b * KPAD_ + r) * CTXD_;
  if (r >= CTX_) {  // padded key rows -> zeros
    for (int i = threadIdx.x; i < CTXD_; i += 256) orow[i] = (_Float16)0.f;
    return;
  }
  const float* irow = ctx + ((size_t)b * CTX_ + r) * CTXD_;
  float s = 0.f, sq = 0.f;
  for (int i = threadIdx.x; i < CTXD_; i += 256) {
    float v = irow[i];
    s += v;
    sq += v * v;
  }
  red[threadIdx.x] = s;
  red[256 + threadIdx.x] = sq;
  __syncthreads();
  for (int o = 128; o > 0; o >>= 1) {
    if (threadIdx.x < o) {
      red[threadIdx.x] += red[threadIdx.x + o];
      red[256 + threadIdx.x] += red[256 + threadIdx.x + o];
    }
    __syncthreads();
  }
  float mean = red[0] / (float)CTXD_;
  float var  = red[256] / (float)CTXD_ - mean * mean;
  float rs   = rsqrtf(var + EPS_);
  for (int i = threadIdx.x; i < CTXD_; i += 256)
    orow[i] = (_Float16)((irow[i] - mean) * rs * lng[i] + lnb[i]);
}

// ------------------------------- kernel 3 ----------------------------------
// One block per (b, head, {K|V}); 4 waves; out = ctx16[96x768] @ W[:,h*32:+32].
// K stored row-major [key][hd] (== N-major B for Q@K^T), scale folded.
// V stored transposed [hd][key]  (== N-major B for P@V).
__global__ __launch_bounds__(128) void kv_gemm_kernel(const _Float16* __restrict__ ctx16,
                                                      const float* __restrict__ Wk,
                                                      const float* __restrict__ Wv,
                                                      _Float16* __restrict__ kmat,
                                                      _Float16* __restrict__ vT) {
  int bid = blockIdx.x;
  int z = bid & 1;          // 0 = K, 1 = V
  int h = (bid >> 1) & 7;
  int b = bid >> 4;
  int wave = threadIdx.x >> 5, lane = threadIdx.x & 31;
  const float* W = z ? Wv : Wk;
  const _Float16* A = ctx16 + (size_t)b * KPAD_ * CTXD_;

  for (int mt = wave; mt < KPAD_ / 16; mt += 4) {
    v8f acc0 = {}, acc1 = {};
    for (int ks = 0; ks < CTXD_ / 32; ++ks) {
      v16h a  = load_a_frag(A + (size_t)(mt * 16) * CTXD_ + ks * 32, CTXD_, lane);
      v16h b0 = load_b_frag_f32(W + (size_t)(ks * 32) * C_ + h * HD_,      C_, lane);
      v16h b1 = load_b_frag_f32(W + (size_t)(ks * 32) * C_ + h * HD_ + 16, C_, lane);
      acc0 = WMMA_F16(a, b0, acc0);
      acc1 = WMMA_F16(a, b1, acc1);
    }
    int n15 = lane & 15, mb = (lane < 16) ? 0 : 8;
    if (z == 0) {
      _Float16* dst = kmat + (size_t)(b * H_ + h) * KPAD_ * HD_;  // [key][hd]
#pragma unroll
      for (int j = 0; j < 8; ++j) {
        int m = mt * 16 + mb + j;  // key index
        dst[(size_t)m * HD_ + n15]      = (_Float16)(acc0[j] * SCALE_);
        dst[(size_t)m * HD_ + 16 + n15] = (_Float16)(acc1[j] * SCALE_);
      }
    } else {
      _Float16* dst = vT + (size_t)(b * H_ + h) * HD_ * KPAD_;    // [hd][key]
#pragma unroll
      for (int j = 0; j < 8; ++j) {
        int m = mt * 16 + mb + j;  // key index
        dst[(size_t)(n15)      * KPAD_ + m] = (_Float16)acc0[j];
        dst[(size_t)(16 + n15) * KPAD_ + m] = (_Float16)acc1[j];
      }
    }
  }
}

// ------------------------------- kernel 4 ----------------------------------
// Transpose+convert Wq/Wo (f32 row-major [k][n]) -> f16 N-major [n][k].
__global__ __launch_bounds__(256) void wcvt_kernel(const float* __restrict__ Wq,
                                                   const float* __restrict__ Wo,
                                                   _Float16* __restrict__ WqT,
                                                   _Float16* __restrict__ WoT) {
  __shared__ float tile[64][65];
  int mat = blockIdx.x >> 4;
  int tr  = (blockIdx.x >> 2) & 3, tc = blockIdx.x & 3;
  const float* W = mat ? Wo : Wq;
  _Float16* WT = mat ? WoT : WqT;
  for (int i = threadIdx.x; i < 64 * 64; i += 256) {
    int r = i >> 6, c = i & 63;
    tile[r][c] = W[(size_t)(tr * 64 + r) * C_ + tc * 64 + c];
  }
  __syncthreads();
  for (int i = threadIdx.x; i < 64 * 64; i += 256) {
    int r = i >> 6, c = i & 63;
    WT[(size_t)(tc * 64 + r) * C_ + tr * 64 + c] = (_Float16)tile[c][r];
  }
}

// ------------------------------- kernel 5 ----------------------------------
__global__ __launch_bounds__(256) void attn_kernel(
    const float* __restrict__ x, const float* __restrict__ gn_g,
    const float* __restrict__ gn_b, const _Float16* __restrict__ WqT,
    const _Float16* __restrict__ WoT, const float* __restrict__ bo,
    const float* __restrict__ gmean, const float* __restrict__ grstd,
    const _Float16* __restrict__ kmat, const _Float16* __restrict__ vT,
    float* __restrict__ out) {
  extern __shared__ char smem[];
  _Float16* sA  = (_Float16*)smem;           // [MT_][LDA_]
  _Float16* sQ  = sA + MT_ * LDA_;           // [MT_][LDA_]
  _Float16* sP  = sQ + MT_ * LDA_;           // [MT_][LDP_]
  _Float16* sBt = sP + MT_ * LDP_;           // [64][LDT_]  weight chunk, N-major
  float*    sS  = (float*)(sBt + 64 * LDT_); // [MT_][LDP_]

  int tid = threadIdx.x, wave = tid >> 5, lane = tid & 31;
  int b  = blockIdx.x >> 8;
  int m0 = (blockIdx.x & 255) * MT_;
  int m0w = wave * 16;
  int n15 = lane & 15, mb = (lane < 16) ? 0 : 8;

  // ---- step 0: x tile -> GroupNorm-normalize -> f16 LDS [m][c]
  const float* xb = x + (size_t)b * C_ * S_;
  for (int i = tid; i < C_ * MT_ / 4; i += 256) {
    int c = i >> 5, mq = i & 31;  // 32 float4 per channel row
    int g = c >> 5;
    float mu = gmean[b * G_ + g], rs = grstd[b * G_ + g];
    float ga = gn_g[c], be = gn_b[c];
    float4 v = *(const float4*)(xb + (size_t)c * S_ + m0 + 4 * mq);
    sA[(size_t)(4 * mq + 0) * LDA_ + c] = (_Float16)((v.x - mu) * rs * ga + be);
    sA[(size_t)(4 * mq + 1) * LDA_ + c] = (_Float16)((v.y - mu) * rs * ga + be);
    sA[(size_t)(4 * mq + 2) * LDA_ + c] = (_Float16)((v.z - mu) * rs * ga + be);
    sA[(size_t)(4 * mq + 3) * LDA_ + c] = (_Float16)((v.w - mu) * rs * ga + be);
  }
  __syncthreads();

  // ---- step 1: Q = sA @ Wq  (4 chunks of 64 N-cols; WqT staged N-major)
  for (int nc = 0; nc < 4; ++nc) {
#if USE_ASYNC_STAGE
    for (int i = tid; i < 64 * 32; i += 256) {  // 64 rows x 32 x 16B
      int n = i >> 5, c16 = i & 31;
      async_copy16(WqT + (size_t)(nc * 64 + n) * C_ + c16 * 8,
                   sBt + (size_t)n * LDT_ + c16 * 8);
    }
    __builtin_amdgcn_s_wait_asynccnt(0);
#else
    for (int i = tid; i < 64 * 32; i += 256) {
      int n = i >> 5, c16 = i & 31;
      *(v8h*)(sBt + (size_t)n * LDT_ + c16 * 8) =
          *(const v8h*)(WqT + (size_t)(nc * 64 + n) * C_ + c16 * 8);
    }
#endif
    __syncthreads();
    v8f acc[4] = {};
#pragma unroll
    for (int ks = 0; ks < 8; ++ks) {
      v16h a = load_a_frag(sA + (size_t)m0w * LDA_ + ks * 32, LDA_, lane);
#pragma unroll
      for (int t = 0; t < 4; ++t) {
        v16h bf = load_b_frag_t(sBt + (size_t)(t * 16) * LDT_ + ks * 32, LDT_, lane);
        acc[t] = WMMA_F16(a, bf, acc[t]);
      }
    }
#pragma unroll
    for (int t = 0; t < 4; ++t)
      store_d_f16(sQ + (size_t)m0w * LDA_ + nc * 64 + t * 16, LDA_, lane, acc[t]);
    __syncthreads();
  }

  // ---- step 2: attention per head (wave-private 16 rows; no barriers)
  const _Float16* kb_ = kmat + (size_t)b * H_ * KPAD_ * HD_;
  const _Float16* vb_ = vT   + (size_t)b * H_ * HD_ * KPAD_;
  for (int h = 0; h < H_; ++h) {
    // scores[16x96] = Q_h[16x32] @ K^T (K stored [key][hd] == N-major B)
    v8f sc[6] = {};
    v16h aq = load_a_frag(sQ + (size_t)m0w * LDA_ + h * HD_, LDA_, lane);
    const _Float16* kh = kb_ + (size_t)h * KPAD_ * HD_;
#pragma unroll
    for (int t = 0; t < 6; ++t) {
      v16h bk = load_b_frag_t(kh + (size_t)(t * 16) * HD_, HD_, lane);
      sc[t] = WMMA_F16(aq, bk, sc[t]);
    }
#pragma unroll
    for (int t = 0; t < 6; ++t)
#pragma unroll
      for (int j = 0; j < 8; ++j)
        sS[(size_t)(m0w + mb + j) * LDP_ + t * 16 + n15] = sc[t][j];

    // row softmax over the 77 valid keys; padded cols -> P = 0
    if (lane < 16) {
      int row = m0w + lane;
      float*    srow = sS + (size_t)row * LDP_;
      _Float16* prow = sP + (size_t)row * LDP_;
      float mx = -3.0e38f;
      for (int kk = 0; kk < CTX_; ++kk) mx = fmaxf(mx, srow[kk]);
      float sum = 0.f;
      for (int kk = 0; kk < CTX_; ++kk) {
        float e = __expf(srow[kk] - mx);
        sum += e;
        prow[kk] = (_Float16)e;
      }
      float inv = 1.f / sum;
      for (int kk = 0; kk < CTX_; ++kk)
        prow[kk] = (_Float16)((float)prow[kk] * inv);
      for (int kk = CTX_; kk < KPAD_; ++kk) prow[kk] = (_Float16)0.f;
    }

    // O_h[16x32] = P[16x96] @ V  (V stored [hd][key] == N-major B)
    v8f oc[2] = {};
    const _Float16* vh = vb_ + (size_t)h * HD_ * KPAD_;
#pragma unroll
    for (int ks = 0; ks < 3; ++ks) {
      v16h ap = load_a_frag(sP + (size_t)m0w * LDP_ + ks * 32, LDP_, lane);
#pragma unroll
      for (int t = 0; t < 2; ++t) {
        v16h bv = load_b_frag_t(vh + (size_t)(t * 16) * KPAD_ + ks * 32, KPAD_, lane);
        oc[t] = WMMA_F16(ap, bv, oc[t]);
      }
    }
#pragma unroll
    for (int t = 0; t < 2; ++t)
      store_d_f16(sA + (size_t)m0w * LDA_ + h * HD_ + t * 16, LDA_, lane, oc[t]);
  }

  // ---- step 3: out = sA @ Wo + bo + residual, stored [b][c][m]
  for (int nc = 0; nc < 4; ++nc) {
    __syncthreads();  // everyone done with step 2 / previous sBt
#if USE_ASYNC_STAGE
    for (int i = tid; i < 64 * 32; i += 256) {
      int n = i >> 5, c16 = i & 31;
      async_copy16(WoT + (size_t)(nc * 64 + n) * C_ + c16 * 8,
                   sBt + (size_t)n * LDT_ + c16 * 8);
    }
    __builtin_amdgcn_s_wait_asynccnt(0);
#else
    for (int i = tid; i < 64 * 32; i += 256) {
      int n = i >> 5, c16 = i & 31;
      *(v8h*)(sBt + (size_t)n * LDT_ + c16 * 8) =
          *(const v8h*)(WoT + (size_t)(nc * 64 + n) * C_ + c16 * 8);
    }
#endif
    __syncthreads();
    v8f acc[4] = {};
#pragma unroll
    for (int ks = 0; ks < 8; ++ks) {
      v16h a = load_a_frag(sA + (size_t)m0w * LDA_ + ks * 32, LDA_, lane);
#pragma unroll
      for (int t = 0; t < 4; ++t) {
        v16h bf = load_b_frag_t(sBt + (size_t)(t * 16) * LDT_ + ks * 32, LDT_, lane);
        acc[t] = WMMA_F16(a, bf, acc[t]);
      }
    }
#pragma unroll
    for (int t = 0; t < 4; ++t) {
      int ccol   = nc * 64 + t * 16 + n15;
      float bias = bo[ccol];
      size_t base = ((size_t)b * C_ + ccol) * S_ + (size_t)(m0 + m0w + mb);
      float4 r0 = *(const float4*)(x + base);
      float4 r1 = *(const float4*)(x + base + 4);
      float4 o0, o1;
      o0.x = acc[t][0] + bias + r0.x;  o0.y = acc[t][1] + bias + r0.y;
      o0.z = acc[t][2] + bias + r0.z;  o0.w = acc[t][3] + bias + r0.w;
      o1.x = acc[t][4] + bias + r1.x;  o1.y = acc[t][5] + bias + r1.y;
      o1.z = acc[t][6] + bias + r1.z;  o1.w = acc[t][7] + bias + r1.w;
      *(float4*)(out + base)     = o0;
      *(float4*)(out + base + 4) = o1;
    }
  }
}

// ------------------------------- launcher ----------------------------------
extern "C" void kernel_launch(void* const* d_in, const int* in_sizes, int n_in,
                              void* d_out, int out_size, void* d_ws, size_t ws_size,
                              hipStream_t stream) {
  (void)in_sizes; (void)n_in; (void)out_size; (void)ws_size;
  const float* x    = (const float*)d_in[0];
  const float* ctx  = (const float*)d_in[1];
  const float* gn_g = (const float*)d_in[2];
  const float* gn_b = (const float*)d_in[3];
  const float* ln_g = (const float*)d_in[4];
  const float* ln_b = (const float*)d_in[5];
  const float* Wq   = (const float*)d_in[6];
  const float* Wk   = (const float*)d_in[7];
  const float* Wv   = (const float*)d_in[8];
  const float* Wo   = (const float*)d_in[9];
  const float* bo   = (const float*)d_in[10];
  float* out = (float*)d_out;

  // workspace: GN stats | ctx16 | K | V^T | WqT | WoT   (~0.76 MB)
  float*    gmean = (float*)d_ws;
  float*    grstd = gmean + 16;
  _Float16* ctx16 = (_Float16*)((char*)d_ws + 256);
  _Float16* kmat  = ctx16 + (size_t)B_ * KPAD_ * CTXD_;
  _Float16* vT    = kmat  + (size_t)B_ * H_ * KPAD_ * HD_;
  _Float16* WqT   = vT    + (size_t)B_ * H_ * HD_ * KPAD_;
  _Float16* WoT   = WqT   + (size_t)C_ * C_;

  gn_stats_kernel<<<B_ * G_, 256, 0, stream>>>(x, gmean, grstd);
  kv_ln_kernel<<<B_ * KPAD_, 256, 0, stream>>>(ctx, ln_g, ln_b, ctx16);
  kv_gemm_kernel<<<B_ * H_ * 2, 128, 0, stream>>>(ctx16, Wk, Wv, kmat, vT);
  wcvt_kernel<<<32, 256, 0, stream>>>(Wq, Wo, WqT, WoT);

  (void)hipFuncSetAttribute((const void*)attn_kernel,
                            hipFuncAttributeMaxDynamicSharedMemorySize, (int)SMEM_ATTN);
  attn_kernel<<<B_ * (S_ / MT_), 256, SMEM_ATTN, stream>>>(
      x, gn_g, gn_b, WqT, WoT, bo, gmean, grstd, kmat, vT, out);
}
